// HeavyEncoderLayer_old_74388833566994
// MI455X (gfx1250) — matrix operand order repaired and checked by reading, not compiled, not run on hardware
//
#include <hip/hip_runtime.h>
#include <hip/hip_bf16.h>
#include <math.h>

typedef __attribute__((ext_vector_type(16))) _Float16 v16h;
typedef __attribute__((ext_vector_type(8)))  float    v8f;
typedef __attribute__((ext_vector_type(2)))  float    v2f;

#define DEV __device__ __forceinline__

constexpr int kN = 50000;
constexpr int kE = 1000000;
constexpr int kH = 25000;

constexpr float INV3    = 0.57735026918962576f;   // 1/sqrt(3)
constexpr float NORMMSG = 0.10206207261596575f;   // 1/sqrt(96)
constexpr float NORMGS  = 0.25f;                  // 1/sqrt(16)
constexpr float NORMGV  = 0.17677669529663687f;   // 1/sqrt(32)
constexpr float NORMH0  = 0.027950849718747374f;  // 1/sqrt(1280)
constexpr float NORMH1  = 0.03125f;               // 1/sqrt(1024)

// ---- workspace layout (float offsets) ----
constexpr size_t OFF_NODEMSG = 0;                                 // kN*112
constexpr size_t OFF_DEG     = OFF_NODEMSG + (size_t)kN * 112;    // kN
constexpr size_t OFF_HSSUM   = OFF_DEG + (size_t)kN;              // kH*16
constexpr size_t OFF_HVSUM   = OFF_HSSUM + (size_t)kH * 16;       // kH*96
constexpr size_t OFF_CNT     = OFF_HVSUM + (size_t)kH * 96;       // kH
constexpr size_t ZERO_END    = OFF_CNT + (size_t)kH;
constexpr size_t OFF_XAS     = ZERO_END;                          // kN*16
constexpr size_t OFF_XAV     = OFF_XAS + (size_t)kN * 16;         // kN*96
constexpr size_t OFF_H0      = OFF_XAV + (size_t)kN * 96;         // kH*16
constexpr size_t OFF_H1      = OFF_H0 + (size_t)kH * 16;          // kH*96

// ---- WMMA wrappers ----
DEV v8f wmma16(v16h a, v16h b, v8f c) {
  return __builtin_amdgcn_wmma_f32_16x16x32_f16(false, a, false, b, (short)0, c,
                                                false, false);
}
DEV v8f wmma4(v2f a, v2f b, v8f c) {
  return __builtin_amdgcn_wmma_f32_16x16x4_f32(false, a, false, b, (short)0, c,
                                               false, false);
}

// B-fragment (f16, K-chunk of 32) for a row-major weight W[K_total][ncols].
// Lane holds column n = colBase + (lane&15); lane-half splits K at granularity 8 (table order).
DEV v16h bfrag_f16(const float* __restrict__ W, int ncols, int kBase,
                   int colBase, int lane) {
  const int half_ = lane >> 4;
  const int n = colBase + (lane & 15);
  v16h b;
#pragma unroll
  for (int j = 0; j < 8; ++j) {
    int k = kBase + ((j < 4) ? (8 * half_ + 2 * j) : (16 + 8 * half_ + 2 * (j - 4)));
    b[2 * j]     = (_Float16)W[(size_t)k * ncols + n];
    b[2 * j + 1] = (_Float16)W[(size_t)(k + 1) * ncols + n];
  }
  return b;
}

// B-fragment (f32, K-chunk of 4) for a row-major weight.
DEV v2f bfrag_f32(const float* __restrict__ W, int ncols, int kb, int colBase,
                  int lane) {
  const int half_ = lane >> 4;
  const int n = colBase + (lane & 15);
  v2f b;
  b[0] = W[(size_t)(kb + 2 * half_) * ncols + n];
  b[1] = W[(size_t)(kb + 2 * half_ + 1) * ncols + n];
  return b;
}

// ---------------------------------------------------------------- init
__global__ void hel_init_kernel(float* __restrict__ ws, size_t cnt) {
  size_t i = (size_t)blockIdx.x * blockDim.x + threadIdx.x;
  if (i < cnt) ws[i] = 0.0f;
}

// ---------------------------------------------------------------- edge phase
// One wave = 16 edges. Edge GEMMs via v_wmma_f32_16x16x32_f16, fused atomic
// scatter into node_msg (L2-resident, 22 MB).
__global__ __launch_bounds__(256) void hel_edge_kernel(
    const float* __restrict__ x, const int* __restrict__ ei,
    const float* __restrict__ ea, const float* __restrict__ W1,
    const float* __restrict__ W2, const float* __restrict__ W3,
    const float* __restrict__ W4, float* __restrict__ node_msg,
    float* __restrict__ deg) {
  __shared__ int   s_src[8][16];
  __shared__ int   s_dst[8][16];
  __shared__ float s_se[8][16];
  __shared__ float s_ve[8][16][3];

  const int wave = threadIdx.x >> 5;
  const int lane = threadIdx.x & 31;
  const long base = (long)blockIdx.x * 128 + (long)wave * 16;

  if (lane < 16) {
    long e = base + lane;
    int srcI = 0, dstI = 0;
    float se = 0.f, w0 = 0.f, w1 = 0.f, w2 = 0.f;
    if (e < kE) {
      srcI = ei[e];
      dstI = ei[(size_t)kE + e];
      se = ea[e * 4 + 0];
      w0 = ea[e * 4 + 1];
      w1 = ea[e * 4 + 2];
      w2 = ea[e * 4 + 3];
      atomicAdd(&deg[dstI], 1.0f);
    }
    s_src[wave][lane] = srcI;
    s_dst[wave][lane] = dstI;
    s_se[wave][lane] = se;
    s_ve[wave][lane][0] = w0;
    s_ve[wave][lane][1] = w1;
    s_ve[wave][lane][2] = w2;
  }
  __syncthreads();

  const int half_ = lane >> 4;
  const int m = lane & 15;  // A-row this lane feeds
  const int n = lane & 15;  // C/D column this lane owns

  // Weight B-fragments (hoisted; L2-resident weights)
  const v16h W1f0 = bfrag_f16(W1, 16, 0, 0, lane);
  const v16h W1f1 = bfrag_f16(W1, 16, 32, 0, lane);
  const v16h W2f  = bfrag_f16(W2, 16, 0, 0, lane);
  const v16h W3f00 = bfrag_f16(W3, 32, 0, 0, lane);
  const v16h W3f01 = bfrag_f16(W3, 32, 0, 16, lane);
  const v16h W3f10 = bfrag_f16(W3, 32, 32, 0, lane);
  const v16h W3f11 = bfrag_f16(W3, 32, 32, 16, lane);
  const v16h W4f0 = bfrag_f16(W4, 32, 0, 0, lane);
  const v16h W4f1 = bfrag_f16(W4, 32, 0, 16, lane);

  const float* __restrict__ xr = x + (size_t)s_src[wave][m] * 160;
  const float seV = s_se[wave][m];
  const float ve0 = s_ve[wave][m][0];
  const float ve1 = s_ve[wave][m][1];
  const float ve2 = s_ve[wave][m][2];

  // K-offset pattern for the 16-bit A 16x32 fragment
  int koff[8];
#pragma unroll
  for (int j = 0; j < 8; ++j)
    koff[j] = (j < 4) ? (8 * half_ + 2 * j) : (16 + 8 * half_ + 2 * (j - 4));

  v16h aS0, aS1, aSe0, aSe1, aT;
#pragma unroll
  for (int j = 0; j < 8; ++j) {
    const int k0 = koff[j];
    const float x0 = xr[k0], x1 = xr[k0 + 1];
    aS0[2 * j] = (_Float16)x0;
    aS0[2 * j + 1] = (_Float16)x1;
    aSe0[2 * j] = (_Float16)(x0 * seV);
    aSe0[2 * j + 1] = (_Float16)(x1 * seV);
    const int k1 = 32 + k0;
    const float y0 = xr[k1], y1 = xr[k1 + 1];
    aS1[2 * j] = (_Float16)y0;
    aS1[2 * j + 1] = (_Float16)y1;
    aSe1[2 * j] = (_Float16)(y0 * seV);
    aSe1[2 * j + 1] = (_Float16)(y1 * seV);
#pragma unroll
    for (int d = 0; d < 2; ++d) {
      const int u = k0 + d;
      const float tv = INV3 * (xr[64 + 3 * u] * ve0 + xr[64 + 3 * u + 1] * ve1 +
                               xr[64 + 3 * u + 2] * ve2);
      aT[2 * j + d] = (_Float16)tv;
    }
  }

  // msg0 = NORM_MSG * ((ss*se)@W1 + (INV3*t)@W2)
  v8f acc0 = {};
  acc0 = wmma16(aSe0, W1f0, acc0);
  acc0 = wmma16(aSe1, W1f1, acc0);
  acc0 = wmma16(aT, W2f, acc0);

  // p = ss@W3 (N=32 as two column halves)
  v8f accP0 = {}, accP1 = {};
  accP0 = wmma16(aS0, W3f00, accP0);
  accP0 = wmma16(aS1, W3f10, accP0);
  accP1 = wmma16(aS0, W3f01, accP1);
  accP1 = wmma16(aS1, W3f11, accP1);

#pragma unroll
  for (int r = 0; r < 8; ++r) {
    const int mm = r + 8 * half_;
    const int dd = s_dst[wave][mm];
    atomicAdd(&node_msg[(size_t)dd * 112 + n], NORMMSG * acc0[r]);
  }

  for (int i = 0; i < 3; ++i) {
    v16h aQ;
#pragma unroll
    for (int j = 0; j < 8; ++j) {
#pragma unroll
      for (int d = 0; d < 2; ++d) {
        const int u = koff[j] + d;
        aQ[2 * j + d] = (_Float16)(xr[64 + 3 * u + i] * seV);
      }
    }
    v8f q0 = {}, q1 = {};
    q0 = wmma16(aQ, W4f0, q0);
    q1 = wmma16(aQ, W4f1, q1);
#pragma unroll
    for (int r = 0; r < 8; ++r) {
      const int mm = r + 8 * half_;
      const int dd = s_dst[wave][mm];
      const float vei = s_ve[wave][mm][i];
      atomicAdd(&node_msg[(size_t)dd * 112 + 16 + n * 3 + i],
                NORMMSG * (accP0[r] * vei + q0[r]));
      atomicAdd(&node_msg[(size_t)dd * 112 + 16 + (16 + n) * 3 + i],
                NORMMSG * (accP1[r] * vei + q1[r]));
    }
  }
}

// ---------------------------------------------------------------- node phase
// Exact f32 WMMA (16x16x4) for ms@Wa, ms@Wb, mv_i@Wc + gating + heavy accum.
__global__ __launch_bounds__(256) void hel_node_kernel(
    const float* __restrict__ node_msg, const float* __restrict__ deg,
    const int* __restrict__ z, const int* __restrict__ canonical,
    const float* __restrict__ Wa, const float* __restrict__ Wb,
    const float* __restrict__ Wc, float* __restrict__ xa_s,
    float* __restrict__ xa_v, float* __restrict__ hs_sum,
    float* __restrict__ hv_sum, float* __restrict__ cnt) {
  __shared__ int   s_node[8][16];   // -1 == invalid row
  __shared__ float s_invd[8][16];
  __shared__ int   s_hvy[8][16];
  __shared__ int   s_can[8][16];

  const int wave = threadIdx.x >> 5;
  const int lane = threadIdx.x & 31;
  const long base = (long)blockIdx.x * 128 + (long)wave * 16;

  if (lane < 16) {
    long nd = base + lane;
    const int ok = nd < kN;
    const int nn = ok ? (int)nd : 0;
    s_invd[wave][lane] = 1.0f / fmaxf(deg[nn], 1.0f);
    int hv = 0, cn = 0;
    if (ok) {
      cn = canonical[nn];
      hv = (z[nn] > 1) ? 1 : 0;
      if (hv) atomicAdd(&cnt[cn], 1.0f);
    }
    s_node[wave][lane] = ok ? nn : -1;
    s_hvy[wave][lane] = hv;
    s_can[wave][lane] = cn;
  }
  __syncthreads();

  const int half_ = lane >> 4;
  const int m = lane & 15;
  const int n = lane & 15;
  const int rnRaw = s_node[wave][m];
  const int rn = rnRaw < 0 ? 0 : rnRaw;
  const float invd = s_invd[wave][m];
  const float* __restrict__ msrow = node_msg + (size_t)rn * 112;

  v8f accA = {}, accB0 = {}, accB1 = {};
#pragma unroll
  for (int kb = 0; kb < 16; kb += 4) {
    v2f a;
    a[0] = msrow[kb + 2 * half_] * invd;
    a[1] = msrow[kb + 2 * half_ + 1] * invd;
    accA = wmma4(a, bfrag_f32(Wa, 16, kb, 0, lane), accA);
    accB0 = wmma4(a, bfrag_f32(Wb, 32, kb, 0, lane), accB0);
    accB1 = wmma4(a, bfrag_f32(Wb, 32, kb, 16, lane), accB1);
  }

  float xs[8], th0[8], th1[8];
#pragma unroll
  for (int r = 0; r < 8; ++r) {
    xs[r] = 1.0f / (1.0f + expf(-NORMGS * accA[r]));
    th0[r] = tanhf(NORMGS * accB0[r]);
    th1[r] = tanhf(NORMGS * accB1[r]);
  }

#pragma unroll
  for (int r = 0; r < 8; ++r) {
    const int mm = r + 8 * half_;
    const int node = s_node[wave][mm];
    if (node >= 0) {
      xa_s[(size_t)node * 16 + n] = xs[r];
      if (s_hvy[wave][mm])
        atomicAdd(&hs_sum[(size_t)s_can[wave][mm] * 16 + n], xs[r]);
    }
  }

  for (int i = 0; i < 3; ++i) {
    v8f accC0 = {}, accC1 = {};
#pragma unroll
    for (int kb = 0; kb < 32; kb += 4) {
      v2f a;
      const int k0 = kb + 2 * half_;
      a[0] = msrow[16 + k0 * 3 + i] * invd;
      a[1] = msrow[16 + (k0 + 1) * 3 + i] * invd;
      accC0 = wmma4(a, bfrag_f32(Wc, 32, kb, 0, lane), accC0);
      accC1 = wmma4(a, bfrag_f32(Wc, 32, kb, 16, lane), accC1);
    }
#pragma unroll
    for (int r = 0; r < 8; ++r) {
      const int mm = r + 8 * half_;
      const int node = s_node[wave][mm];
      if (node >= 0) {
        const float xv0 = th0[r] * NORMGV * accC0[r];
        const float xv1 = th1[r] * NORMGV * accC1[r];
        xa_v[(size_t)node * 96 + n * 3 + i] = xv0;
        xa_v[(size_t)node * 96 + (16 + n) * 3 + i] = xv1;
        if (s_hvy[wave][mm]) {
          const size_t cb = (size_t)s_can[wave][mm] * 96;
          atomicAdd(&hv_sum[cb + n * 3 + i], xv0);
          atomicAdd(&hv_sum[cb + (16 + n) * 3 + i], xv1);
        }
      }
    }
  }
}

// ---------------------------------------------------------------- heavy phase
// Bilinear forms as GEMMs over outer-product features: h0 K=1280 N=16,
// h1 K=1024 N=32 per spatial i. Exact f32 WMMA, LDS-staged hs/hv tiles.
__global__ __launch_bounds__(128) void hel_heavy_kernel(
    const float* __restrict__ hs_sum, const float* __restrict__ hv_sum,
    const float* __restrict__ cnt, const float* __restrict__ Wss,
    const float* __restrict__ Wvv, const float* __restrict__ Wsv,
    const float* __restrict__ Wvs, float* __restrict__ h0o,
    float* __restrict__ h1o) {
  __shared__ float s_hs[4][16][16];
  __shared__ float s_hv[4][16][96];
  __shared__ int   s_idx[4][16];

  const int wave = threadIdx.x >> 5;
  const int lane = threadIdx.x & 31;
  const long base = (long)blockIdx.x * 64 + (long)wave * 16;

  if (lane < 16) {
    long h = base + lane;
    const int ok = h < kH;
    const int hh = ok ? (int)h : 0;
    const float invc = ok ? (1.0f / fmaxf(cnt[hh], 1.0f)) : 0.0f;
    for (int k2 = 0; k2 < 16; ++k2)
      s_hs[wave][lane][k2] = hs_sum[(size_t)hh * 16 + k2] * invc;
    for (int k2 = 0; k2 < 96; ++k2)
      s_hv[wave][lane][k2] = hv_sum[(size_t)hh * 96 + k2] * invc;
    s_idx[wave][lane] = ok ? hh : -1;
  }
  __syncthreads();

  const int half_ = lane >> 4;
  const int m = lane & 15;
  const int n = lane & 15;
  const float* __restrict__ hs = s_hs[wave][m];
  const float* __restrict__ hv = s_hv[wave][m];

  // h0: K = 256 (hs⊗hs vs Wss) + 1024 (INV3*hvGram vs Wvv)
  v8f acc = {};
#pragma unroll 4
  for (int kb = 0; kb < 1280; kb += 4) {
    v2f a, b;
#pragma unroll
    for (int j = 0; j < 2; ++j) {
      const int k = kb + 2 * half_ + j;
      float av, bv;
      if (k < 256) {
        const int u = k >> 4, v = k & 15;
        av = hs[u] * hs[v];
        bv = Wss[(size_t)k * 16 + n];
      } else {
        const int kk = k - 256;
        const int u = kk >> 5, v = kk & 31;
        av = INV3 * (hv[u * 3] * hv[v * 3] + hv[u * 3 + 1] * hv[v * 3 + 1] +
                     hv[u * 3 + 2] * hv[v * 3 + 2]);
        bv = Wvv[(size_t)kk * 16 + n];
      }
      a[j] = av;
      b[j] = bv;
    }
    acc = wmma4(a, b, acc);
  }
#pragma unroll
  for (int r = 0; r < 8; ++r) {
    const int mm = r + 8 * half_;
    const int hh = s_idx[wave][mm];
    if (hh >= 0) h0o[(size_t)hh * 16 + n] = NORMH0 * acc[r];
  }

  // h1 per spatial component i: K = 512 (hs⊗hv_i vs Wsv) + 512 (hv_i⊗hs vs Wvs)
  for (int i = 0; i < 3; ++i) {
    v8f a0 = {}, a1 = {};
#pragma unroll 2
    for (int kb = 0; kb < 1024; kb += 4) {
      v2f a, b0, b1;
#pragma unroll
      for (int j = 0; j < 2; ++j) {
        const int k = kb + 2 * half_ + j;
        float av, bv0, bv1;
        if (k < 512) {
          const int u = k >> 5, v = k & 31;
          av = hs[u] * hv[v * 3 + i];
          bv0 = Wsv[(size_t)k * 32 + n];
          bv1 = Wsv[(size_t)k * 32 + 16 + n];
        } else {
          const int kk = k - 512;
          const int u = kk >> 4, v = kk & 15;
          av = hv[u * 3 + i] * hs[v];
          bv0 = Wvs[(size_t)kk * 32 + n];
          bv1 = Wvs[(size_t)kk * 32 + 16 + n];
        }
        a[j] = av;
        b0[j] = bv0;
        b1[j] = bv1;
      }
      a0 = wmma4(a, b0, a0);
      a1 = wmma4(a, b1, a1);
    }
#pragma unroll
    for (int r = 0; r < 8; ++r) {
      const int mm = r + 8 * half_;
      const int hh = s_idx[wave][mm];
      if (hh >= 0) {
        h1o[(size_t)hh * 96 + n * 3 + i] = NORMH1 * a0[r];
        h1o[(size_t)hh * 96 + (16 + n) * 3 + i] = NORMH1 * a1[r];
      }
    }
  }
}

// ---------------------------------------------------------------- final select
__global__ void hel_final_kernel(const int* __restrict__ z,
                                 const int* __restrict__ canonical,
                                 const float* __restrict__ xa_s,
                                 const float* __restrict__ xa_v,
                                 const float* __restrict__ h0o,
                                 const float* __restrict__ h1o,
                                 float* __restrict__ out) {
  size_t i = (size_t)blockIdx.x * blockDim.x + threadIdx.x;
  if (i >= (size_t)kN * 112) return;
  const int nIdx = (int)(i / 112);
  const int c = (int)(i % 112);
  const bool heavy = z[nIdx] > 1;
  float val;
  if (c < 16) {
    val = heavy ? h0o[(size_t)canonical[nIdx] * 16 + c]
                : xa_s[(size_t)nIdx * 16 + c];
  } else {
    const int cc = c - 16;
    val = heavy ? h1o[(size_t)canonical[nIdx] * 96 + cc]
                : xa_v[(size_t)nIdx * 96 + cc];
  }
  out[i] = val;
}

// ---------------------------------------------------------------- launcher
extern "C" void kernel_launch(void* const* d_in, const int* in_sizes, int n_in,
                              void* d_out, int out_size, void* d_ws,
                              size_t ws_size, hipStream_t stream) {
  (void)in_sizes; (void)n_in; (void)out_size; (void)ws_size;
  const float* x   = (const float*)d_in[0];
  const int*   ei  = (const int*)d_in[1];
  const float* ea  = (const float*)d_in[2];
  const int*   z   = (const int*)d_in[3];
  const int*   can = (const int*)d_in[4];
  const float* W1  = (const float*)d_in[5];
  const float* W2  = (const float*)d_in[6];
  const float* W3  = (const float*)d_in[7];
  const float* W4  = (const float*)d_in[8];
  const float* Wa  = (const float*)d_in[9];
  const float* Wb  = (const float*)d_in[10];
  const float* Wc  = (const float*)d_in[11];
  const float* Wss = (const float*)d_in[12];
  const float* Wvv = (const float*)d_in[13];
  const float* Wsv = (const float*)d_in[14];
  const float* Wvs = (const float*)d_in[15];

  float* ws = (float*)d_ws;
  float* node_msg = ws + OFF_NODEMSG;
  float* deg      = ws + OFF_DEG;
  float* hs_sum   = ws + OFF_HSSUM;
  float* hv_sum   = ws + OFF_HVSUM;
  float* cnt      = ws + OFF_CNT;
  float* xa_s     = ws + OFF_XAS;
  float* xa_v     = ws + OFF_XAV;
  float* h0o      = ws + OFF_H0;
  float* h1o      = ws + OFF_H1;
  float* out = (float*)d_out;

  const size_t zcount = ZERO_END;
  hel_init_kernel<<<(unsigned)((zcount + 255) / 256), 256, 0, stream>>>(ws,
                                                                        zcount);

  const unsigned edgeBlocks = (unsigned)((kE + 127) / 128);
  hel_edge_kernel<<<edgeBlocks, 256, 0, stream>>>(x, ei, ea, W1, W2, W3, W4,
                                                  node_msg, deg);

  const unsigned nodeBlocks = (unsigned)((kN + 127) / 128);
  hel_node_kernel<<<nodeBlocks, 256, 0, stream>>>(node_msg, deg, z, can, Wa, Wb,
                                                  Wc, xa_s, xa_v, hs_sum,
                                                  hv_sum, cnt);

  const unsigned heavyBlocks = (unsigned)((kH + 63) / 64);
  hel_heavy_kernel<<<heavyBlocks, 128, 0, stream>>>(hs_sum, hv_sum, cnt, Wss,
                                                    Wvv, Wsv, Wvs, h0o, h1o);

  const size_t total = (size_t)kN * 112;
  hel_final_kernel<<<(unsigned)((total + 255) / 256), 256, 0, stream>>>(
      z, can, xa_s, xa_v, h0o, h1o, out);
}
